// FCNRZ_37967510897303
// MI455X (gfx1250) — matrix operand order repaired and checked by reading, not compiled, not run on hardware
//
#include <hip/hip_runtime.h>

// ---------------------------------------------------------------------------
// ROBE-Z hashed 3-layer MLP: out = (relu(relu(x@W0)@W1))@W2
//   W_l[k][n] = hw_l[(k*A + n*B + C) & (S-1)]      (S is a power of two)
//
// CDNA5 structure:
//   * activations bf16 (intermediates stored bf16 by the epilogue)
//   * A tiles: Tensor Data Mover global->LDS (one issue per WG per K-step,
//     TDM pad gives 80B row stride for aligned b128 fragment reads)
//   * B tiles: hash-gather from L2-resident hw; indices computed
//     independently (S | 2^32 so "(base + kk*A) & mask" is exact)
//   * 2-deep software pipeline over double-buffered LDS: TDM + gather loads
//     for tile t+1 overlap the 8 WMMAs of tile t; one barrier per K-step
//   * GEMM on v_wmma_f32_16x16x32_bf16, f32 accumulate, fused ReLU epilogue
// ---------------------------------------------------------------------------

typedef __attribute__((ext_vector_type(16))) __bf16 v16bf;
typedef __attribute__((ext_vector_type(8)))  __bf16 bf16x8;
typedef __attribute__((ext_vector_type(8)))  float  v8f;
typedef __attribute__((ext_vector_type(4)))  float  f32x4;
typedef __attribute__((ext_vector_type(4)))  unsigned int u32x4;
typedef __attribute__((ext_vector_type(8)))  int    i32x8;
typedef __attribute__((ext_vector_type(4)))  int    i32x4;

#define BLOCK_M 128
#define BLOCK_N 128
#define BLOCK_K 32
#define LDA 40                       // bf16/row: 64B data + 16B TDM pad = 80B
#define LDB 40                       // B stored transposed [n][k]
#define A_BYTES   (BLOCK_M * LDA * 2)              // 10240
#define B_BYTES   (BLOCK_N * LDB * 2)              // 10240
#define LDS_TOTAL (2 * A_BYTES + 2 * B_BYTES)      // 40960 (double buffered)

// one-time f32 -> bf16 conversion of the network input
__global__ void f32_to_bf16_kernel(const float* __restrict__ src,
                                   __bf16* __restrict__ dst, int n) {
  int i = (blockIdx.x * blockDim.x + threadIdx.x) * 8;
  if (i + 8 <= n) {
    f32x4 a = *(const f32x4*)&src[i];
    f32x4 b = *(const f32x4*)&src[i + 4];
    bf16x8 o;
#pragma unroll
    for (int e = 0; e < 4; ++e) { o[e] = (__bf16)a[e]; o[e + 4] = (__bf16)b[e]; }
    *(bf16x8*)&dst[i] = o;
  }
}

// issue one TDM transfer: 32(K) x 128(M) bf16 tile -> LDS with 16B/row pad
__device__ __forceinline__ void issue_tdm_A(unsigned ldsAddr,
                                            unsigned long long ga, int K) {
  u32x4 g0;
  g0[0] = 1u;                                    // count=1, no gather mode
  g0[1] = ldsAddr;                               // lds_addr (bytes)
  g0[2] = (unsigned)(ga & 0xFFFFFFFFull);        // global_addr[31:0]
  g0[3] = (unsigned)((ga >> 32) & 0x01FFFFFFull) | (2u << 30); // [56:32], type=2
  i32x8 g1;
  // data_size=2B(code1), pad_enable, pad_interval=16DW(3), pad_amount=4DW(3)
  g1[0] = (int)((1u << 16) | (1u << 20) | (3u << 22) | (3u << 25));
  g1[1] = (int)(((unsigned)K & 0xFFFFu) << 16);                   // tensor_dim0 lo
  g1[2] = (int)((((unsigned)K >> 16) & 0xFFFFu) | (2048u << 16)); // hi | tensor_dim1
  g1[3] = (int)(((unsigned)BLOCK_K) << 16);                       // tile_dim0=32
  g1[4] = (int)(BLOCK_M);                                         // tile_dim1=128
  g1[5] = (int)K;                                                 // dim0_stride lo
  g1[6] = 0;
  g1[7] = 0;
  i32x4 gz4 = {0, 0, 0, 0};
  i32x8 gz8 = {0, 0, 0, 0, 0, 0, 0, 0};
  __builtin_amdgcn_tensor_load_to_lds(g0, g1, gz4, gz4, gz8, 0);
}

// independent-index hash gather: 16 L2-resident loads, no dependency chain
__device__ __forceinline__ void gather_B(const float* __restrict__ hw,
                                         unsigned base, unsigned hashA,
                                         unsigned smask, float v[16]) {
#pragma unroll
  for (int kk = 0; kk < 16; ++kk)
    v[kk] = hw[(base + (unsigned)kk * hashA) & smask];
}

__device__ __forceinline__ void store_B(__bf16* Bbuf, int bn, int bk0,
                                        const float v[16]) {
  bf16x8 lo8, hi8;
#pragma unroll
  for (int e = 0; e < 8; ++e) { lo8[e] = (__bf16)v[e]; hi8[e] = (__bf16)v[8 + e]; }
  bf16x8* dst = (bf16x8*)&Bbuf[bn * LDB + bk0];
  dst[0] = lo8;
  dst[1] = hi8;
}

__global__ __launch_bounds__(256)
void robez_mlp_layer(const __bf16* __restrict__ in,   // [2048, K] bf16 row-major
                     const float*  __restrict__ hw,   // hashed weight store
                     void* __restrict__ outp,         // [2048, N]
                     int K, int N, unsigned smask,
                     unsigned hashA, unsigned hashB, unsigned hashC,
                     int relu_bf16_out)               // 1: bf16+ReLU, 0: f32
{
  extern __shared__ __align__(16) char smem[];
  // layout: Abuf0 | Abuf1 | Bbuf0 | Bbuf1  (pointers computed at runtime to
  // avoid constant aggregate initializers with LDS addrspacecasts)

  const int tid  = threadIdx.x;
  const int lane = tid & 31;
  const int wid  = tid >> 5;      // 8 waves
  const int wm   = wid >> 2;      // 0..1 -> 64 M-rows per wave
  const int wn   = wid & 3;       // 0..3 -> 32 N-cols per wave

  const int m0 = blockIdx.y * BLOCK_M;
  const int n0 = blockIdx.x * BLOCK_N;

  v8f acc[4][2];
#pragma unroll
  for (int i = 0; i < 4; ++i)
#pragma unroll
    for (int j = 0; j < 2; ++j)
#pragma unroll
      for (int e = 0; e < 8; ++e) acc[i][j][e] = 0.0f;

  // B-tile generation: thread -> (column n, half of K range)
  const int bn  = tid & 127;
  const int bk0 = (tid >> 7) << 4;            // 0 or 16
  // hash base for (k = bk0, n = n0+bn); u32 wraparound is exact mod S
  const unsigned hbase0 = (unsigned)bk0 * hashA + (unsigned)(n0 + bn) * hashB + hashC;
  const unsigned hstepK = (unsigned)BLOCK_K * hashA;   // advance per K-step

  const unsigned ldsA0 = (unsigned)(size_t)(void*)smem;
  const unsigned long long gbase =
      (unsigned long long)(size_t)in + (unsigned long long)m0 * (unsigned)K * 2ull;

  const int T = K / BLOCK_K;

  // ---- prologue: fill buffer 0 with tile 0 ----
  if (wid == 0) issue_tdm_A(ldsA0, gbase, K);
  {
    float v[16];
    gather_B(hw, hbase0, hashA, smask, v);
    store_B((__bf16*)(smem + 2 * A_BYTES), bn, bk0, v);
  }
  if (wid == 0) __builtin_amdgcn_s_wait_tensorcnt(0);
  __syncthreads();

  for (int t = 0; t < T; ++t) {
    const int p = t & 1;

    // ---- kick off tile t+1 into the other buffer (overlaps WMMAs below) ----
    float vnext[16];
    const bool have_next = (t + 1 < T);
    if (have_next) {
      if (wid == 0)
        issue_tdm_A(ldsA0 + (unsigned)((p ^ 1) * A_BYTES),
                    gbase + (unsigned long long)(t + 1) * (BLOCK_K * 2), K);
      gather_B(hw, hbase0 + (unsigned)(t + 1) * hstepK, hashA, smask, vnext);
    }

    // ---- compute tile t: fragments per ISA 7.12.2 16-bit layouts, 8 WMMAs ----
    const __bf16* At = (const __bf16*)(smem + p * A_BYTES);
    const __bf16* Bt = (const __bf16*)(smem + 2 * A_BYTES + p * B_BYTES);
    v16bf afrag[4], bfrag[2];
    const int koffA = (lane < 16) ? 0 : 8;   // A: lanes 0-15 K0-7/16-23, 16-31 K8-15/24-31
#pragma unroll
    for (int i = 0; i < 4; ++i) {
      int m = (wm << 6) + (i << 4) + (lane & 15);
      bf16x8 lo = *(const bf16x8*)&At[m * LDA + koffA];
      bf16x8 hi = *(const bf16x8*)&At[m * LDA + koffA + 16];
      afrag[i] = __builtin_shufflevector(lo, hi, 0,1,2,3,4,5,6,7,8,9,10,11,12,13,14,15);
    }
    const int koffB = (lane < 16) ? 0 : 8;   // B: lanes 0-15 K0-7/16-23, 16-31 K8-15/24-31
#pragma unroll
    for (int j = 0; j < 2; ++j) {
      int n = (wn << 5) + (j << 4) + (lane & 15);
      bf16x8 lo = *(const bf16x8*)&Bt[n * LDB + koffB];
      bf16x8 hi = *(const bf16x8*)&Bt[n * LDB + koffB + 16];
      bfrag[j] = __builtin_shufflevector(lo, hi, 0,1,2,3,4,5,6,7,8,9,10,11,12,13,14,15);
    }
#pragma unroll
    for (int i = 0; i < 4; ++i)
#pragma unroll
      for (int j = 0; j < 2; ++j)
        acc[i][j] = __builtin_amdgcn_wmma_f32_16x16x32_bf16(
            false, afrag[i], false, bfrag[j], (short)0, acc[i][j], false, false);

    // ---- finish staging tile t+1, then one barrier per K-step ----
    if (have_next) {
      store_B((__bf16*)(smem + 2 * A_BYTES + (p ^ 1) * B_BYTES), bn, bk0, vnext);
      if (wid == 0) __builtin_amdgcn_s_wait_tensorcnt(0);
      __syncthreads();
    }
  }

  // ---- epilogue: C/D layout VGPR r -> M = r + (lane<16?0:8), N = lane&15 ----
  const int nlo = lane & 15;
  const int mhi = (lane < 16) ? 0 : 8;
#pragma unroll
  for (int i = 0; i < 4; ++i) {
#pragma unroll
    for (int j = 0; j < 2; ++j) {
      int gm_base = m0 + (wm << 6) + (i << 4) + mhi;
      int gn      = n0 + (wn << 5) + (j << 4) + nlo;
      if (relu_bf16_out) {
        __bf16* out = (__bf16*)outp;
#pragma unroll
        for (int r = 0; r < 8; ++r)
          out[(size_t)(gm_base + r) * N + gn] = (__bf16)fmaxf(acc[i][j][r], 0.0f);
      } else {
        float* out = (float*)outp;
#pragma unroll
        for (int r = 0; r < 8; ++r)
          out[(size_t)(gm_base + r) * N + gn] = acc[i][j][r];
      }
    }
  }
}

extern "C" void kernel_launch(void* const* d_in, const int* in_sizes, int n_in,
                              void* d_out, int out_size, void* d_ws, size_t ws_size,
                              hipStream_t stream) {
  (void)in_sizes; (void)n_in; (void)out_size; (void)ws_size;

  const float* x   = (const float*)d_in[0];   // [2048, 1024]
  const float* hw0 = (const float*)d_in[1];   // 2^20
  const float* hw1 = (const float*)d_in[2];   // 2^20
  const float* hw2 = (const float*)d_in[3];   // 2^22
  float* out = (float*)d_out;                 // [2048, 32000] f32

  // workspace: xb (4MB) | h1 bf16 (16MB) | h2 bf16 (16MB)
  __bf16* xb = (__bf16*)d_ws;
  __bf16* h1 = (__bf16*)((char*)d_ws + (size_t)2048 * 1024 * 2);
  __bf16* h2 = (__bf16*)((char*)d_ws + (size_t)2048 * 1024 * 2 + (size_t)2048 * 4096 * 2);

  dim3 blk(256);
  const int GM = 2048 / BLOCK_M;              // 16

  f32_to_bf16_kernel<<<(2048 * 1024 / 8) / 256, blk, 0, stream>>>(x, xb, 2048 * 1024);

  robez_mlp_layer<<<dim3(4096 / BLOCK_N, GM), blk, LDS_TOTAL, stream>>>(
      xb, hw0, h1, 1024, 4096, (1u << 20) - 1u, 9973u, 31013u, 557u, 1);
  robez_mlp_layer<<<dim3(4096 / BLOCK_N, GM), blk, LDS_TOTAL, stream>>>(
      h1, hw1, h2, 4096, 4096, (1u << 20) - 1u, 10007u, 31019u, 563u, 1);
  robez_mlp_layer<<<dim3(32000 / BLOCK_N, GM), blk, LDS_TOTAL, stream>>>(
      h2, hw2, out, 4096, 32000, (1u << 22) - 1u, 10039u, 31039u, 569u, 0);
}
